// BVIFormer_56178172232287
// MI455X (gfx1250) — matrix operand order repaired and checked by reading, not compiled, not run on hardware
//
#include <hip/hip_runtime.h>
#include <hip/hip_bf16.h>
#include <math.h>

// ---------------------------------------------------------------------------
// CDNA5 / gfx1250 implementation of the BVIFormer local+pool attention block.
// All dense GEMMs ride v_wmma_f32_16x16x32_f16 (f16 in, f32 accumulate).
// GEMM waves compute a 16x64 strip (4 WMMA tiles sharing one A fragment).
// Attention is fused flash-style per 16-row q tile: logits -> softmax in LDS
// -> P@V via WMMA, never materializing the (B,h,N,205) attention in HBM.
// All contiguous fragment loads use float4 (global_load_b128).
// ---------------------------------------------------------------------------

typedef _Float16 f16;
typedef __attribute__((ext_vector_type(16))) _Float16 v16h;
typedef __attribute__((ext_vector_type(8)))  float    v8f;

#define DEVFN __device__ __forceinline__
#define NEG_INF (-__builtin_inff())

constexpr int BB    = 4;
constexpr int IH    = 56;
constexpr int IW    = 56;
constexpr int NN    = IH * IW;        // 3136
constexpr int CC    = 256;
constexpr int HEADS = 8;
constexpr int HD    = 32;
constexpr int LL    = 9;
constexpr int SRF   = 4;
constexpr int PH    = IH / SRF;       // 14
constexpr int PW    = IW / SRF;       // 14
constexpr int PL    = PH * PW;        // 196
constexpr int KTAB  = 2048;
constexpr int CPBH  = 512;
constexpr int ATT_W = LL + PL;        // 205
constexpr int SLW   = 224;            // padded logits row stride in LDS

// ---------------------------------------------------------------------------
// WMMA fragment loaders (contiguous k order, same for A and B)
// A 16x32 f16: lane l<16 -> row l, K elems 0..15 contiguous; lane l+16 -> K 16..31
// B 32x16 f16: lane n<16 -> col n, K elems 0..15 contiguous; lane n+16 -> K 16..31
// C/D 16x16 f32: VGPR i: lanes 0-15 row i, lanes 16-31 row i+8; col = lane%16
// ---------------------------------------------------------------------------

DEVFN void cvt4(v16h& a, int off, float4 v) {
    a[off + 0] = (f16)v.x; a[off + 1] = (f16)v.y;
    a[off + 2] = (f16)v.z; a[off + 3] = (f16)v.w;
}

// A fragment from a row-major f32 matrix (base 32B-aligned runs of 8 floats)
DEVFN v16h load_a_frag(const float* __restrict__ A, int lda, int m0, int kk, int lane) {
    int lr = lane & 15, hs = lane >> 4;
    const float* row = A + (size_t)(m0 + lr) * lda + kk;
    const float* p0 = row + (hs ? 16 : 0);
    const float* p1 = row + (hs ? 24 : 8);
    v16h a;
    cvt4(a, 0,  *(const float4*)(p0));
    cvt4(a, 4,  *(const float4*)(p0 + 4));
    cvt4(a, 8,  *(const float4*)(p1));
    cvt4(a, 12, *(const float4*)(p1 + 4));
    return a;
}

// B fragment from W rows (row-major n_out x K); 16 contiguous K per lane
DEVFN v16h load_b_frag(const float* __restrict__ W, int ldw, int n0, int kk, int lane) {
    int lr = lane & 15, hs = lane >> 4;
    const float* p = W + (size_t)(n0 + lr) * ldw + kk + (hs ? 16 : 0);
    v16h b;
    cvt4(b, 0,  *(const float4*)(p));
    cvt4(b, 4,  *(const float4*)(p + 4));
    cvt4(b, 8,  *(const float4*)(p + 8));
    cvt4(b, 12, *(const float4*)(p + 12));
    return b;
}

// ---------------------------------------------------------------------------
// Generic GEMM:  Out[m, coff+n] = act( sum_k A[m,k] * W[n,k] + bias[n] )
// One wave per 16x64 strip (4 WMMA tiles reuse one A fragment per K step).
// M multiple of 16; N multiple of 64; K multiple of 32.
// ---------------------------------------------------------------------------
template <int ACT>  // 0 = none, 1 = exact GELU
__global__ __launch_bounds__(32)
void gemm_wmma_kernel(const float* __restrict__ A, const float* __restrict__ W,
                      const float* __restrict__ bias, float* __restrict__ Out,
                      int K, int ldo, int col_off) {
    int lane = threadIdx.x;
    int m0 = blockIdx.x * 16;
    int n0 = blockIdx.y * 64;
    v8f acc[4] = {v8f{}, v8f{}, v8f{}, v8f{}};
    for (int kk = 0; kk < K; kk += 32) {
        v16h a = load_a_frag(A, K, m0, kk, lane);
#pragma unroll
        for (int t = 0; t < 4; ++t) {
            v16h b = load_b_frag(W, K, n0 + t * 16, kk, lane);
            acc[t] = __builtin_amdgcn_wmma_f32_16x16x32_f16(false, a, false, b, (short)0,
                                                            acc[t], false, false);
        }
    }
    int lr = lane & 15, hs = lane >> 4;
#pragma unroll
    for (int t = 0; t < 4; ++t) {
        int col = n0 + t * 16 + lr;
        float bv = bias[col];
#pragma unroll
        for (int i = 0; i < 8; ++i) {
            int row = m0 + i + (hs ? 8 : 0);
            float v = acc[t][i] + bv;
            if (ACT == 1) v = 0.5f * v * (1.f + erff(v * 0.7071067811865475f));
            Out[(size_t)row * ldo + col_off + col] = v;
        }
    }
}

// ---------------------------------------------------------------------------
// CPB MLP table: tab[k][h] = cpb2_w[h] . relu(ct[k] @ cpb1_w.T + cpb1_b) + cpb2_b[h]
// ---------------------------------------------------------------------------
__global__ void cpb_kernel(const float* __restrict__ ct, const float* __restrict__ w1,
                           const float* __restrict__ b1, const float* __restrict__ w2,
                           const float* __restrict__ b2, float* __restrict__ tab) {
    int k = blockIdx.x * blockDim.x + threadIdx.x;
    if (k >= KTAB) return;
    float c0 = ct[2 * k], c1 = ct[2 * k + 1];
    float acc[HEADS];
#pragma unroll
    for (int h = 0; h < HEADS; ++h) acc[h] = b2[h];
    for (int j = 0; j < CPBH; ++j) {
        float hv = fmaxf(c0 * w1[2 * j] + c1 * w1[2 * j + 1] + b1[j], 0.f);
#pragma unroll
        for (int h = 0; h < HEADS; ++h) acc[h] += w2[h * CPBH + j] * hv;
    }
#pragma unroll
    for (int h = 0; h < HEADS; ++h) tab[k * HEADS + h] = acc[h];
}

// ---------------------------------------------------------------------------
// Per-row prep: q_norm, q_scaled, and in-place L2 normalization of k_loc.
// Block = one (b,n) row, 256 threads = 8 waves = 8 heads, lane = channel.
// ---------------------------------------------------------------------------
__global__ __launch_bounds__(256)
void prep_kernel(float* __restrict__ qkv, float* __restrict__ qn, float* __restrict__ qs,
                 const float* __restrict__ seq, const float* __restrict__ temp,
                 const float* __restrict__ qemb) {
    int row = blockIdx.x;          // b*NN + n
    int n = row % NN;
    int t = threadIdx.x;
    int h = t >> 5, d = t & 31;
    float* qrow = qkv + (size_t)row * (3 * CC);

    float q = qrow[h * HD + d];
    float ss = q * q;
#pragma unroll
    for (int m = 16; m >= 1; m >>= 1) ss += __shfl_xor(ss, m, 32);
    float qnv = q / fmaxf(sqrtf(ss), 1e-12f);
    float sp = logf(1.f + expf(temp[h]));               // softplus(temperature[h])
    float qsv = (qnv + qemb[h * HD + d]) * sp * seq[n];
    qn[(size_t)row * CC + h * HD + d] = qnv;
    qs[(size_t)row * CC + h * HD + d] = qsv;

    float kv = qrow[CC + h * HD + d];
    float ks = kv * kv;
#pragma unroll
    for (int m = 16; m >= 1; m >>= 1) ks += __shfl_xor(ks, m, 32);
    qrow[CC + h * HD + d] = kv / fmaxf(sqrtf(ks), 1e-12f);
}

// ---------------------------------------------------------------------------
// 4x4 mean pool of gelu(sr) + LayerNorm over channels. Block = (b,p), 256 thr.
// ---------------------------------------------------------------------------
__global__ __launch_bounds__(256)
void pool_ln_kernel(const float* __restrict__ xsr, const float* __restrict__ g,
                    const float* __restrict__ bt, float* __restrict__ xpool) {
    int bp = blockIdx.x;           // b*PL + p
    int b = bp / PL, p = bp % PL;
    int py = p / PW, px = p % PW;
    int c = threadIdx.x;
    float s = 0.f;
#pragma unroll
    for (int dy = 0; dy < SRF; ++dy)
#pragma unroll
        for (int dx = 0; dx < SRF; ++dx) {
            int pix = (py * SRF + dy) * IW + (px * SRF + dx);
            s += xsr[((size_t)b * NN + pix) * CC + c];
        }
    float v = s * (1.f / 16.f);
    __shared__ float red[256];
    red[c] = v; __syncthreads();
    for (int st = 128; st >= 1; st >>= 1) { if (c < st) red[c] += red[c + st]; __syncthreads(); }
    float mu = red[0] * (1.f / CC);
    __syncthreads();
    red[c] = (v - mu) * (v - mu); __syncthreads();
    for (int st = 128; st >= 1; st >>= 1) { if (c < st) red[c] += red[c + st]; __syncthreads(); }
    float var = red[0] * (1.f / CC);
    xpool[(size_t)bp * CC + c] = (v - mu) * rsqrtf(var + 1e-5f) * g[c] + bt[c];
}

// ---------------------------------------------------------------------------
// In-place L2 normalization of k_pool heads. Block = (b,p), 256 thr = 8 heads.
// ---------------------------------------------------------------------------
__global__ __launch_bounds__(256)
void kpool_norm_kernel(float* __restrict__ kvp) {
    int bp = blockIdx.x;
    int t = threadIdx.x;
    int h = t >> 5, d = t & 31;
    float* rowp = kvp + (size_t)bp * (2 * CC);
    float k = rowp[h * HD + d];
    float ss = k * k;
#pragma unroll
    for (int m = 16; m >= 1; m >>= 1) ss += __shfl_xor(ss, m, 32);
    rowp[h * HD + d] = k / fmaxf(sqrtf(ss), 1e-12f);
}

// ---------------------------------------------------------------------------
// Fused attention: one wave per (b, h, 16-row q tile).
//   local logits (9, VALU) + pool logits (196, WMMA K=32) -> softmax in LDS
//   -> x_glob = P @ v_pool (WMMA, K padded to 224) + local AV (VALU gathers)
// ---------------------------------------------------------------------------
__global__ __launch_bounds__(32)
void attn_fused_kernel(const float* __restrict__ qs, const float* __restrict__ qn,
                       const float* __restrict__ qkv, const float* __restrict__ kvp,
                       const float* __restrict__ tab, const int* __restrict__ ridx,
                       const float* __restrict__ rpb, const float* __restrict__ ltok,
                       const float* __restrict__ lbias, float* __restrict__ attout) {
    __shared__ float sL[16][SLW];   // logits -> probabilities (cols 0..8 local, 9.. pool)
    __shared__ float sT[16][16];    // q_norm @ learnable_tokens
    __shared__ float sO[16][HD];    // x_glob

    int lane = threadIdx.x;
    int n0 = blockIdx.x * 16;
    int h  = blockIdx.y;
    int b  = blockIdx.z;
    int lr = lane & 15, hs = lane >> 4;
    int o0 = hs ? 16 : 0, o1 = hs ? 24 : 8;
    const size_t qoff = (size_t)b * NN * CC + h * HD;

    // ---- A fragment: q_scaled tile (16 x 32), K = head dim (one WMMA per tile)
    v16h aq = load_a_frag(qs + qoff, CC, n0, 0, lane);

    // ---- pooled logits: 13 tiles of 16 pool keys (196 -> 208 padded)
    for (int j = 0; j < 13; ++j) {
        int p0 = j * 16;
        int p = p0 + lr;
        v16h bk;
        if (p < PL) {
            const float* base = kvp + ((size_t)b * PL + p) * (2 * CC) + h * HD + (hs ? 16 : 0);
            cvt4(bk, 0,  *(const float4*)(base));
            cvt4(bk, 4,  *(const float4*)(base + 4));
            cvt4(bk, 8,  *(const float4*)(base + 8));
            cvt4(bk, 12, *(const float4*)(base + 12));
        } else {
#pragma unroll
            for (int e = 0; e < 16; ++e) bk[e] = (f16)0.f;
        }
        v8f c = {};
        c = __builtin_amdgcn_wmma_f32_16x16x32_f16(false, aq, false, bk, (short)0, c, false, false);
#pragma unroll
        for (int i = 0; i < 8; ++i) {
            int row = i + (hs ? 8 : 0);
            float v;
            if (p < PL) {
                int n = n0 + row;
                v = c[i] + tab[(size_t)ridx[(size_t)n * PL + p] * HEADS + h];
            } else {
                v = NEG_INF;
            }
            sL[row][LL + p0 + lr] = v;
        }
    }

    // ---- token term: q_norm (16x32) @ learnable_tokens (32x9, padded to 16 cols)
    {
        v16h an = load_a_frag(qn + qoff, CC, n0, 0, lane);
        v16h bt;
        {
            int l = lr;
            int koff = hs ? 16 : 0;
#pragma unroll
            for (int e = 0; e < 16; ++e)
                bt[e] = (f16)((l < LL) ? ltok[((size_t)h * HD + koff + e) * LL + l] : 0.f);
        }
        v8f c = {};
        c = __builtin_amdgcn_wmma_f32_16x16x32_f16(false, an, false, bt, (short)0, c, false, false);
#pragma unroll
        for (int i = 0; i < 8; ++i) sT[i + (hs ? 8 : 0)][lr] = c[i];
    }

    // ---- local logits (9 taps, padding mask computed analytically)
    for (int t = lane; t < 16 * LL; t += 32) {
        int row = t / LL, l = t % LL;
        int n = n0 + row;
        int y = n / IW, x = n % IW;
        int ny = y + l / 3 - 1, nx = x + l % 3 - 1;
        float v;
        if (ny >= 0 && ny < IH && nx >= 0 && nx < IW) {
            const float4* q4 = (const float4*)(qs + qoff + (size_t)n * CC);
            const float4* k4 = (const float4*)(qkv + ((size_t)b * NN + ny * IW + nx) * (3 * CC)
                                               + CC + h * HD);
            float acc = 0.f;
#pragma unroll
            for (int dd = 0; dd < 8; ++dd) {
                float4 qv = q4[dd], kv = k4[dd];
                acc += qv.x * kv.x + qv.y * kv.y + qv.z * kv.z + qv.w * kv.w;
            }
            v = acc + rpb[h * LL + l];
        } else {
            v = NEG_INF;
        }
        sL[row][l] = v;
    }
    __syncthreads();

    // ---- softmax over 205 cols; two lanes per row (split column ranges)
    {
        int row = lr;
        int c0 = hs ? 104 : 0;
        int c1 = hs ? ATT_W : 104;
        float mx = NEG_INF;
        for (int ci = c0; ci < c1; ++ci) mx = fmaxf(mx, sL[row][ci]);
        mx = fmaxf(mx, __shfl_xor(mx, 16, 32));
        float sm = 0.f;
        for (int ci = c0; ci < c1; ++ci) sm += expf(sL[row][ci] - mx);
        sm += __shfl_xor(sm, 16, 32);
        float inv = 1.f / sm;
        for (int ci = c0; ci < c1; ++ci) sL[row][ci] = expf(sL[row][ci] - mx) * inv;
    }
    __syncthreads();

    // ---- x_glob = P(16x196) @ v_pool(196x32), K padded to 224 (guarded)
    v8f cg0 = {}, cg1 = {};
    for (int kk = 0; kk < 224; kk += 32) {
        v16h ap;
#pragma unroll
        for (int e = 0; e < 8; ++e) {
            int k0 = kk + o0 + e, k1 = kk + o1 + e;
            ap[e]     = (f16)((k0 < PL) ? sL[lr][LL + k0] : 0.f);
            ap[8 + e] = (f16)((k1 < PL) ? sL[lr][LL + k1] : 0.f);
        }
#pragma unroll
        for (int nt = 0; nt < 2; ++nt) {
            v16h bv;
            int d = nt * 16 + lr;
            int koff = kk + (hs ? 16 : 0);
#pragma unroll
            for (int e = 0; e < 16; ++e) {
                int k = koff + e;
                bv[e] = (f16)((k < PL) ? kvp[((size_t)b * PL + k) * (2 * CC) + CC + h * HD + d] : 0.f);
            }
            if (nt == 0)
                cg0 = __builtin_amdgcn_wmma_f32_16x16x32_f16(false, ap, false, bv, (short)0, cg0, false, false);
            else
                cg1 = __builtin_amdgcn_wmma_f32_16x16x32_f16(false, ap, false, bv, (short)0, cg1, false, false);
        }
    }
#pragma unroll
    for (int i = 0; i < 8; ++i) {
        sO[i + (hs ? 8 : 0)][lr]      = cg0[i];
        sO[i + (hs ? 8 : 0)][16 + lr] = cg1[i];
    }
    __syncthreads();

    // ---- local AV + write-out: lane = channel d
    for (int row = 0; row < 16; ++row) {
        int n = n0 + row;
        int y = n / IW, x = n % IW;
        float acc = sO[row][lane];
#pragma unroll
        for (int l = 0; l < LL; ++l) {
            int ny = y + l / 3 - 1, nx = x + l % 3 - 1;
            if (ny >= 0 && ny < IH && nx >= 0 && nx < IW) {
                float w = sT[row][l] + lbias[h * LL + l] + sL[row][l];
                float v = qkv[((size_t)b * NN + ny * IW + nx) * (3 * CC) + 2 * CC + h * HD + lane];
                acc += w * v;
            }
        }
        attout[((size_t)b * NN + n) * CC + h * HD + lane] = acc;
    }
}

// ---------------------------------------------------------------------------
// Host-side launcher
// ---------------------------------------------------------------------------
extern "C" void kernel_launch(void* const* d_in, const int* in_sizes, int n_in,
                              void* d_out, int out_size, void* d_ws, size_t ws_size,
                              hipStream_t stream) {
    (void)in_sizes; (void)n_in; (void)out_size;

    const float* x      = (const float*)d_in[0];
    const int*   ridx   = (const int*)d_in[3];
    const float* ct     = (const float*)d_in[4];
    const float* seq    = (const float*)d_in[5];
    const float* q_w    = (const float*)d_in[7];
    const float* q_b    = (const float*)d_in[8];
    const float* kv_w   = (const float*)d_in[9];
    const float* kv_b   = (const float*)d_in[10];
    const float* sr_w   = (const float*)d_in[11];
    const float* sr_b   = (const float*)d_in[12];
    const float* norm_g = (const float*)d_in[13];
    const float* norm_b = (const float*)d_in[14];
    const float* cpb1_w = (const float*)d_in[15];
    const float* cpb1_b = (const float*)d_in[16];
    const float* cpb2_w = (const float*)d_in[17];
    const float* cpb2_b = (const float*)d_in[18];
    const float* temp   = (const float*)d_in[19];
    const float* qemb   = (const float*)d_in[20];
    const float* rpb    = (const float*)d_in[21];
    const float* ltok   = (const float*)d_in[22];
    const float* lbias  = (const float*)d_in[23];
    const float* proj_w = (const float*)d_in[24];
    const float* proj_b = (const float*)d_in[25];

    float* ws = (float*)d_ws;
    size_t o = 0;
    float* qkv    = ws + o; o += (size_t)BB * NN * (3 * CC);   // q|k|v rows
    float* qn     = ws + o; o += (size_t)BB * NN * CC;         // q_norm
    float* qsd    = ws + o; o += (size_t)BB * NN * CC;         // q_scaled
    float* xsr    = ws + o; o += (size_t)BB * NN * CC;         // gelu(sr)
    float* xpool  = ws + o; o += (size_t)BB * PL * CC;         // pooled + LN
    float* kvp    = ws + o; o += (size_t)BB * PL * (2 * CC);   // k_pool|v_pool
    float* tab    = ws + o; o += (size_t)KTAB * HEADS;         // CPB table
    float* attout = ws + o; o += (size_t)BB * NN * CC;         // pre-proj output
    if (ws_size < o * sizeof(float)) return;                   // scratch guard

    const int MROWS = BB * NN;          // 12544, multiple of 16

    // 1. CPB MLP table
    cpb_kernel<<<KTAB / 256, 256, 0, stream>>>(ct, cpb1_w, cpb1_b, cpb2_w, cpb2_b, tab);

    // 2. QKV projections (fused output buffer, q cols 0..255, kv cols 256..767)
    gemm_wmma_kernel<0><<<dim3(MROWS / 16, CC / 64), 32, 0, stream>>>(
        x, q_w, q_b, qkv, CC, 3 * CC, 0);
    gemm_wmma_kernel<0><<<dim3(MROWS / 16, 2 * CC / 64), 32, 0, stream>>>(
        x, kv_w, kv_b, qkv, CC, 3 * CC, CC);

    // 3. SR 1x1 conv + exact GELU
    gemm_wmma_kernel<1><<<dim3(MROWS / 16, CC / 64), 32, 0, stream>>>(
        x, sr_w, sr_b, xsr, CC, CC, 0);

    // 4. q_norm / q_scaled / k_loc normalization
    prep_kernel<<<BB * NN, 256, 0, stream>>>(qkv, qn, qsd, seq, temp, qemb);

    // 5. 4x4 pool + LayerNorm
    pool_ln_kernel<<<BB * PL, 256, 0, stream>>>(xsr, norm_g, norm_b, xpool);

    // 6. pooled KV projection
    gemm_wmma_kernel<0><<<dim3(BB * PL / 16, 2 * CC / 64), 32, 0, stream>>>(
        xpool, kv_w, kv_b, kvp, CC, 2 * CC, 0);

    // 7. k_pool L2 normalization
    kpool_norm_kernel<<<BB * PL, 256, 0, stream>>>(kvp);

    // 8. fused local+pool attention
    attn_fused_kernel<<<dim3(NN / 16, HEADS, BB), 32, 0, stream>>>(
        qsd, qn, qkv, kvp, tab, ridx, rpb, ltok, lbias, attout);

    // 9. output projection -> d_out
    gemm_wmma_kernel<0><<<dim3(MROWS / 16, CC / 64), 32, 0, stream>>>(
        attout, proj_w, proj_b, (float*)d_out, CC, CC, 0);
}
